// Online_2765958938685
// MI455X (gfx1250) — matrix operand order repaired and checked by reading, not compiled, not run on hardware
//
#include <hip/hip_runtime.h>
#include <hip/hip_bf16.h>

typedef __bf16 bf16_t;
typedef __attribute__((ext_vector_type(16))) __bf16 v16bf;
typedef __attribute__((ext_vector_type(8)))  __bf16 v8bf;
typedef __attribute__((ext_vector_type(8)))  float  v8f;
typedef __attribute__((ext_vector_type(4)))  int    v4i;

typedef __attribute__((address_space(1))) v4i gv4i_t;   // global int4
typedef __attribute__((address_space(3))) v4i lv4i_t;   // LDS int4

#define DIMC  384
#define BATCH 256
#define EPS   1e-5f

// LDS pitch for the staged A tile: 16 rows x 768B of channels, padded to
// 784B/row (196 dwords; 196 % 64 = 4) so rows hit distinct LDS banks.
#define A_PITCH 784
#define A_BYTES (16 * A_PITCH)

__device__ __forceinline__ int clampi(int v, int lo, int hi) {
    return v < lo ? lo : (v > hi ? hi : v);
}

// ---- gfx1250 async global->LDS (ASYNCcnt) ---------------------------------
__device__ __forceinline__ void async_ld16(const void* g, void* l) {
#if __has_builtin(__builtin_amdgcn_global_load_async_to_lds_b128)
    __builtin_amdgcn_global_load_async_to_lds_b128(
        (gv4i_t*)g, (lv4i_t*)l, 0, 0);
#else
    unsigned loff = (unsigned)(size_t)(__attribute__((address_space(3))) char*)(char*)l;
    asm volatile("global_load_async_to_lds_b128 %0, %1, off"
                 :: "v"(loff), "v"(g) : "memory");
#endif
}
__device__ __forceinline__ void wait_async0() {
#if __has_builtin(__builtin_amdgcn_s_wait_asynccnt)
    __builtin_amdgcn_s_wait_asynccnt(0);
#else
    asm volatile("s_wait_asynccnt 0x0" ::: "memory");
#endif
}

// ---------------------------------------------------------------------------
// Stage 0: bbox -> sample indices, gather target rows into a replicate-padded
// (16,16,384) bf16 crop per batch (replicate pad == clamped sample index).
// ---------------------------------------------------------------------------
__global__ void gather_pad_kernel(const float* __restrict__ target,
                                  const float* __restrict__ bbox,
                                  bf16_t* __restrict__ taPad) {
    const int b = blockIdx.x;
    __shared__ int rows_p[16];
    __shared__ int cols_p[16];
    if (threadIdx.x < 16) {
        const float bx = bbox[b * 4 + 0] * 24.f;
        const float by = bbox[b * 4 + 1] * 24.f;
        const float bw = bbox[b * 4 + 2] * 24.f;
        const float bh = bbox[b * 4 + 3] * 24.f;
        const float l0 = 0.5f * (bw + bh);
        const float l  = sqrtf((bw + l0) * (bh + l0));
        const int x_min = clampi((int)ceilf (bx - 0.5f * l), 0, 23);
        const int y_min = clampi((int)ceilf (by - 0.5f * l), 0, 23);
        const int x_max = clampi((int)truncf(bx + 0.5f * l), 0, 23);
        const int y_max = clampi((int)truncf(by + 0.5f * l), 0, 23);
        const int i  = threadIdx.x;            // padded coordinate 0..15
        const int pi = clampi(i - 2, 0, 11);   // replicate-clamped crop index
        rows_p[i] = clampi(y_min + (pi * (y_max - y_min)) / 12, 0, 23);
        cols_p[i] = clampi(x_min + (pi * (x_max - x_min)) / 12, 0, 23);
    }
    __syncthreads();

    const float* tb = target + (size_t)b * 576 * DIMC;
    bf16_t*      ob = taPad  + (size_t)b * 256 * DIMC;
    for (int e = threadIdx.x; e < 256 * DIMC; e += blockDim.x) {
        const int p  = e / DIMC;
        const int c  = e % DIMC;
        const int py = p >> 4, px = p & 15;
        const int idx = rows_p[py] * 24 + cols_p[px];
        ob[e] = (bf16_t)tb[(size_t)idx * DIMC + c];
    }
}

// ---------------------------------------------------------------------------
// Weight repack: (O=384, I=384, TAPS) f32 -> (I, TAPS, O) bf16.
// ---------------------------------------------------------------------------
template <int TAPS>
__global__ void reshape_w_kernel(const float* __restrict__ W,
                                 bf16_t* __restrict__ Wr) {
    const int total = DIMC * DIMC * TAPS;
    for (int e = blockIdx.x * blockDim.x + threadIdx.x; e < total;
         e += gridDim.x * blockDim.x) {
        const int o   = e / (DIMC * TAPS);
        const int rem = e % (DIMC * TAPS);
        const int i   = rem / TAPS;
        const int t   = rem % TAPS;
        Wr[((size_t)i * TAPS + t) * DIMC + o] = (bf16_t)W[e];
    }
}

// ---------------------------------------------------------------------------
// Implicit-GEMM conv on v_wmma_f32_16x16x32_bf16, A staged via async
// global->LDS DMA (double-buffered per tap), B streamed from L2.
//   M = 144 positions (9 tiles), N = 384 out-ch (24 tiles), K = 384*TAPS.
//   Block = 8 waves = one (batch, M-tile); wave w owns N-tiles {3w..3w+2}.
// Per tap: 16 rows x 768B A tile -> LDS (768 b128 async ops, 3/thread),
// then 12 K-steps x 3 WMMA reading A from LDS (ds_load_b128, conflict-free
// via 784B pitch) while the next tap's DMA is in flight.
// ---------------------------------------------------------------------------
template <int TAPS, int KDIM, int HP, int WP, int MODE>
__global__ void conv_wmma_kernel(const bf16_t* __restrict__ Xp,   // (B,HP,WP,384) bf16
                                 const bf16_t* __restrict__ Wr,   // (384,TAPS,384) bf16
                                 const float* __restrict__ gam,
                                 const float* __restrict__ bet,
                                 const float* __restrict__ mu,
                                 const float* __restrict__ var,
                                 bf16_t* __restrict__ OutPad,     // MODE 0
                                 const float* __restrict__ temp,  // MODE 1
                                 float* __restrict__ outF) {      // MODE 1
    __shared__ __attribute__((aligned(16))) char ldsA[2][A_BYTES];

    const int tid  = threadIdx.x;
    const int lane = tid & 31;
    const int wave = tid >> 5;
    const int b    = blockIdx.x / 9;
    const int mt   = blockIdx.x % 9;
    const int mloc = lane & 15;
    const int hi8  = (lane >> 4) << 3;    // 0 or 8

    const char* Xb = (const char*)(Xp + (size_t)b * HP * WP * DIMC);

    // Loader assignment: 768 x b128 chunks per tap, 3 per thread.
    int gBase[3], lOff[3];
#pragma unroll
    for (int k = 0; k < 3; ++k) {
        const int f     = tid + k * 256;       // 0..767
        const int row   = f / 48;              // A-tile row (position)
        const int chunk = f % 48;              // 16B chunk within 768B row
        const int mg = mt * 16 + row;
        const int oy = mg / 12, ox = mg % 12;
        gBase[k] = (oy * WP + ox) * (DIMC * 2) + chunk * 16;  // tap (0,0) bytes
        lOff[k]  = row * A_PITCH + chunk * 16;
    }

    auto issue = [&](int t, int buf) {
        const int ky = t / KDIM, kx = t % KDIM;
        const int tapOff = (ky * WP + kx) * (DIMC * 2);
#pragma unroll
        for (int k = 0; k < 3; ++k)
            async_ld16(Xb + gBase[k] + tapOff, &ldsA[buf][lOff[k]]);
    };

    v8f acc0 = {}, acc1 = {}, acc2 = {};

    issue(0, 0);
    wait_async0();
    __syncthreads();

    for (int t = 0; t < TAPS; ++t) {
        if (t + 1 < TAPS) issue(t + 1, (t + 1) & 1);

        const bf16_t* lrow = (const bf16_t*)(&ldsA[t & 1][mloc * A_PITCH]);
        for (int c0 = 0; c0 < DIMC; c0 += 32) {
            union { v16bf v; v8bf h[2]; } A;
            A.h[0] = *(const v8bf*)(lrow + c0 + hi8);        // K = hi8 + 0..7
            A.h[1] = *(const v8bf*)(lrow + c0 + hi8 + 16);   // K = hi8 + 16..23

            const bf16_t* wk = Wr + ((size_t)(c0 + hi8) * TAPS + t) * DIMC + mloc;
#pragma unroll
            for (int j = 0; j < 3; ++j) {
                const int nt = wave * 3 + j;
                union { v16bf v; v8bf h[2]; } Bv;
#pragma unroll
                for (int s = 0; s < 8; ++s) {
                    Bv.h[0][s] = wk[(size_t)s * TAPS * DIMC + nt * 16];
                    Bv.h[1][s] = wk[(size_t)(s + 16) * TAPS * DIMC + nt * 16];
                }
                v8f& acc = (j == 0) ? acc0 : (j == 1) ? acc1 : acc2;
                acc = __builtin_amdgcn_wmma_f32_16x16x32_bf16(
                          false, A.v, false, Bv.v, (short)0, acc, false, false);
            }
        }
        wait_async0();     // my slice of tap t+1 landed in LDS
        __syncthreads();   // everyone done reading tap t & loading t+1
    }

    // Epilogue: BN (y = x*sc + sh) + leaky ReLU (+ residual / replicate pad).
#pragma unroll
    for (int j = 0; j < 3; ++j) {
        const v8f& acc = (j == 0) ? acc0 : (j == 1) ? acc1 : acc2;
        const int nt  = wave * 3 + j;
        const int n_g = nt * 16 + (lane & 15);
        const float sc = gam[n_g] / sqrtf(var[n_g] + EPS);
        const float sh = bet[n_g] - mu[n_g] * sc;
#pragma unroll
        for (int r = 0; r < 8; ++r) {
            const int ml  = r + hi8;
            const int m_g = mt * 16 + ml;
            float y = acc[r] * sc + sh;
            y = (y > 0.f) ? y : 0.01f * y;
            if constexpr (MODE == 0) {
                const int oy = m_g / 12, ox = m_g % 12;
                bf16_t hv = (bf16_t)y;
                bf16_t* OP = OutPad + (size_t)b * 196 * DIMC;
                auto st = [&](int py, int px) {
                    OP[((size_t)py * 14 + px) * DIMC + n_g] = hv;
                };
                st(oy + 1, ox + 1);
                if (oy == 0)  st(0,  ox + 1);
                if (oy == 11) st(13, ox + 1);
                if (ox == 0)  st(oy + 1, 0);
                if (ox == 11) st(oy + 1, 13);
                if (oy == 0  && ox == 0)  st(0, 0);
                if (oy == 0  && ox == 11) st(0, 13);
                if (oy == 11 && ox == 0)  st(13, 0);
                if (oy == 11 && ox == 11) st(13, 13);
            } else {
                const size_t oi = ((size_t)b * 144 + m_g) * DIMC + n_g;
                outF[oi] = temp[oi] + y;
            }
        }
    }
}

// ---------------------------------------------------------------------------
extern "C" void kernel_launch(void* const* d_in, const int* in_sizes, int n_in,
                              void* d_out, int out_size, void* d_ws, size_t ws_size,
                              hipStream_t stream) {
    const float* temp   = (const float*)d_in[0];   // (256,144,384)
    const float* target = (const float*)d_in[1];   // (256,576,384)
    const float* bbox   = (const float*)d_in[2];   // (256,4)
    const float* W1     = (const float*)d_in[3];   // (384,384,5,5)
    const float* g1     = (const float*)d_in[4];
    const float* b1     = (const float*)d_in[5];
    const float* m1     = (const float*)d_in[6];
    const float* v1     = (const float*)d_in[7];
    const float* W2     = (const float*)d_in[8];   // (384,384,3,3)
    const float* g2     = (const float*)d_in[9];
    const float* b2     = (const float*)d_in[10];
    const float* m2     = (const float*)d_in[11];
    const float* v2     = (const float*)d_in[12];
    float* out = (float*)d_out;

    // Workspace layout (all offsets 256B-aligned).
    char* ws = (char*)d_ws;
    const size_t sz_ta = (size_t)BATCH * 256 * DIMC * sizeof(bf16_t); // 50331648
    const size_t sz_h1 = (size_t)BATCH * 196 * DIMC * sizeof(bf16_t); // 38535168
    const size_t sz_w1 = (size_t)DIMC * 25 * DIMC * sizeof(bf16_t);   //  7372800
    bf16_t* taPad = (bf16_t*)(ws);
    bf16_t* h1Pad = (bf16_t*)(ws + sz_ta);
    bf16_t* W1r   = (bf16_t*)(ws + sz_ta + sz_h1);
    bf16_t* W2r   = (bf16_t*)(ws + sz_ta + sz_h1 + sz_w1);

    gather_pad_kernel<<<BATCH, 256, 0, stream>>>(target, bbox, taPad);

    reshape_w_kernel<25><<<4096, 256, 0, stream>>>(W1, W1r);
    reshape_w_kernel<9><<<4096, 256, 0, stream>>>(W2, W2r);

    // Conv1: 5x5 on 16x16 pad -> BN+leaky -> replicate-padded 14x14 bf16.
    conv_wmma_kernel<25, 5, 16, 16, 0><<<BATCH * 9, 256, 0, stream>>>(
        taPad, W1r, g1, b1, m1, v1, h1Pad, nullptr, nullptr);

    // Conv2: 3x3 on 14x14 pad -> BN+leaky + temp residual -> f32 out.
    conv_wmma_kernel<9, 3, 14, 14, 1><<<BATCH * 9, 256, 0, stream>>>(
        h1Pad, W2r, g2, b2, m2, v2, nullptr, temp, out);
}
